// Net_48576080117975
// MI455X (gfx1250) — matrix operand order repaired and checked by reading, not compiled, not run on hardware
//
#include <hip/hip_runtime.h>
#include <hip/hip_bf16.h>
#include <math.h>

#define B_      2
#define T_      16
#define D_      768
#define L_      4
#define H_      12
#define DK_     64
#define DFF_    3072
#define VOCAB_  32128
#define ENC_    256
#define NBUK_   32

typedef __attribute__((ext_vector_type(16))) __bf16 v16bf;
typedef __attribute__((ext_vector_type(8)))  __bf16 v8bf;
typedef __attribute__((ext_vector_type(8)))  float  v8f;

// ---------------------------------------------------------------- utilities

__global__ __launch_bounds__(256) void cvt_f32_bf16(const float* __restrict__ in,
                                                    __bf16* __restrict__ out, long n) {
    long i = (long)blockIdx.x * blockDim.x + threadIdx.x;
    long stride = (long)gridDim.x * blockDim.x;
    for (; i < n; i += stride) out[i] = (__bf16)in[i];
}

// Transpose [K,N] f32 -> [N,K] bf16, per layer (blockIdx.z). K,N multiples of 32.
__global__ __launch_bounds__(256) void cvt_transpose_bf16(const float* __restrict__ in,
                                                          __bf16* __restrict__ out,
                                                          int K, int N) {
    __shared__ float tile[32][33];
    const float* inl = in + (long)blockIdx.z * K * N;
    __bf16* outl = out + (long)blockIdx.z * K * N;
    int k0 = blockIdx.x * 32, n0 = blockIdx.y * 32;
    int tx = threadIdx.x & 31, ty = threadIdx.x >> 5;      // 32 x 8
    for (int i = ty; i < 32; i += 8)
        tile[i][tx] = inl[(long)(k0 + i) * N + n0 + tx];
    __syncthreads();
    for (int i = ty; i < 32; i += 8)
        outl[(long)(n0 + i) * K + k0 + tx] = (__bf16)tile[tx][i];
}

__global__ void init_tokens(int* tokens) {
    if (threadIdx.x < B_ * T_) tokens[threadIdx.x] = 0;   // START_ID == PAD_ID == 0
}

__global__ __launch_bounds__(256) void embed_kernel(const int* __restrict__ tokens,
                                                    const float* __restrict__ emb,
                                                    float* __restrict__ x) {
    int r = blockIdx.x;                       // r = b*T + t  (32 rows)
    int tok = tokens[r];
    const float* src = emb + (long)tok * D_;
    float* dst = x + (long)r * D_;
    for (int c = threadIdx.x; c < D_; c += 256) dst[c] = src[c];
}

// rmsnorm: out_bf16 = x * rsqrt(mean(x^2)+eps) * w * extra_scale
__global__ __launch_bounds__(256) void rmsnorm_kernel(const float* __restrict__ x,
                                                      const float* __restrict__ w,
                                                      __bf16* __restrict__ out,
                                                      float extra) {
    __shared__ float buf[256];
    int r = blockIdx.x;
    const float* xp = x + (long)r * D_;
    float ss = 0.f;
    for (int c = threadIdx.x; c < D_; c += 256) { float v = xp[c]; ss += v * v; }
    buf[threadIdx.x] = ss;
    __syncthreads();
    for (int s = 128; s > 0; s >>= 1) {
        if (threadIdx.x < s) buf[threadIdx.x] += buf[threadIdx.x + s];
        __syncthreads();
    }
    float scale = rsqrtf(buf[0] / (float)D_ + 1e-6f) * extra;
    __bf16* op = out + (long)r * D_;
    for (int c = threadIdx.x; c < D_; c += 256) op[c] = (__bf16)(xp[c] * scale * w[c]);
}

// ---------------------------------------------------------------- WMMA GEMM
// C[M,N] = A_bf16[M,K] @ Bt_bf16[N,K]^T   (B always stored transposed -> all
// loads are contiguous 2x16B per lane). Each wave computes a 32x32 macro-tile
// (4 accumulators, A/B operands shared across tiles). Block = 4 waves = 32x128.
// M % 32 == 0, N % 128 == 0, K % 32 == 0 -> no bounds checks, EXEC all ones.
template<bool RESID, bool RELU, bool OUT_BF16>
__global__ __launch_bounds__(128) void gemm_wmma(const __bf16* __restrict__ A,
                                                 const __bf16* __restrict__ Bt,
                                                 float* __restrict__ Cf,
                                                 __bf16* __restrict__ Cb,
                                                 const float* __restrict__ X,
                                                 int K, int ldc) {
    const int lane = threadIdx.x & 31;
    const int wave = threadIdx.x >> 5;
    const int m0 = blockIdx.x * 32;
    const int n0 = blockIdx.y * 128 + wave * 32;
    const int r   = lane & 15;
    const int grp = lane >> 4;

    const __bf16* Ar0 = A  + (long)(m0 + r)      * K;
    const __bf16* Ar1 = A  + (long)(m0 + 16 + r) * K;
    const __bf16* Br0 = Bt + (long)(n0 + r)      * K;
    const __bf16* Br1 = Bt + (long)(n0 + 16 + r) * K;

    v8f acc00 = {}, acc01 = {}, acc10 = {}, acc11 = {};
    for (int k0 = 0; k0 < K; k0 += 32) {
        // unconditional speculative prefetch of the weight stream (dropped if invalid)
        __builtin_prefetch(Br0 + k0 + 128, 0, 1);
        __builtin_prefetch(Br1 + k0 + 128, 0, 1);
        // A 16x32 bf16 operand: lanes 0-15 (row r) hold K {0..7,16..23}; 16-31 {8..15,24..31}
        v8bf a0lo = *(const v8bf*)(Ar0 + k0 + grp * 8);
        v8bf a0hi = *(const v8bf*)(Ar0 + k0 + 16 + grp * 8);
        v8bf a1lo = *(const v8bf*)(Ar1 + k0 + grp * 8);
        v8bf a1hi = *(const v8bf*)(Ar1 + k0 + 16 + grp * 8);
        // B 32x16 operand from [N,K]: lane holds col n, K {0..7,16..23} (lanes 0-15) / {8..15,24..31} (16-31)
        v8bf b0lo = *(const v8bf*)(Br0 + k0 + grp * 8);
        v8bf b0hi = *(const v8bf*)(Br0 + k0 + 16 + grp * 8);
        v8bf b1lo = *(const v8bf*)(Br1 + k0 + grp * 8);
        v8bf b1hi = *(const v8bf*)(Br1 + k0 + 16 + grp * 8);

        v16bf a0, a1, b0, b1;
#pragma unroll
        for (int i = 0; i < 8; i++) {
            a0[i] = a0lo[i]; a0[8 + i] = a0hi[i];
            a1[i] = a1lo[i]; a1[8 + i] = a1hi[i];
            b0[i] = b0lo[i]; b0[8 + i] = b0hi[i];
            b1[i] = b1lo[i]; b1[8 + i] = b1hi[i];
        }
        acc00 = __builtin_amdgcn_wmma_f32_16x16x32_bf16(false, a0, false, b0, (short)0, acc00, false, false);
        acc01 = __builtin_amdgcn_wmma_f32_16x16x32_bf16(false, a0, false, b1, (short)0, acc01, false, false);
        acc10 = __builtin_amdgcn_wmma_f32_16x16x32_bf16(false, a1, false, b0, (short)0, acc10, false, false);
        acc11 = __builtin_amdgcn_wmma_f32_16x16x32_bf16(false, a1, false, b1, (short)0, acc11, false, false);
    }

#pragma unroll
    for (int tij = 0; tij < 4; tij++) {
        int ti = tij >> 1, tj = tij & 1;
        v8f acc = (tij == 0) ? acc00 : (tij == 1) ? acc01 : (tij == 2) ? acc10 : acc11;
        int col = n0 + tj * 16 + r;
#pragma unroll
        for (int i = 0; i < 8; i++) {
            int row = m0 + ti * 16 + i + 8 * grp;
            long idx = (long)row * ldc + col;
            float v = acc[i];
            if (RESID) v += X[idx];
            if (RELU)  v = v > 0.f ? v : 0.f;
            if (OUT_BF16) Cb[idx] = (__bf16)v;
            else          Cf[idx] = v;
        }
    }
}

// ---------------------------------------------------------------- attention

// Self attention, T=16, with T5 relative-position bias + causal mask (no 1/sqrt(dk)).
__global__ __launch_bounds__(256) void self_attn_kernel(const float* __restrict__ q,
                                                        const float* __restrict__ k,
                                                        const float* __restrict__ v,
                                                        const float* __restrict__ rel_bias,
                                                        __bf16* __restrict__ out) {
    __shared__ float S[16][16];
    int b = blockIdx.x / H_, h = blockIdx.x % H_;
    int qi = threadIdx.x >> 4, ki = threadIdx.x & 15;

    const float* qp = q + (long)(b * T_ + qi) * D_ + h * DK_;
    const float* kp = k + (long)(b * T_ + ki) * D_ + h * DK_;
    float s = 0.f;
#pragma unroll 8
    for (int d = 0; d < DK_; d++) s += qp[d] * kp[d];

    int rp = qi - ki;
    int bucket;
    if (rp <= 0)       bucket = 0;                 // masked/future -> distance 0
    else if (rp < 16)  bucket = rp;
    else {
        bucket = 16 + (int)(logf((float)rp * (1.0f / 16.0f)) * (16.0f / logf(8.0f)));
        if (bucket > NBUK_ - 1) bucket = NBUK_ - 1;
    }
    s += rel_bias[bucket * H_ + h];
    if (ki > qi) s -= 1e9f;

    S[qi][ki] = s;
    __syncthreads();
    float m = -1e30f;
    for (int j = 0; j < 16; j++) m = fmaxf(m, S[qi][j]);
    float e = expf(s - m);
    __syncthreads();
    S[qi][ki] = e;
    __syncthreads();
    float sum = 0.f;
    for (int j = 0; j < 16; j++) sum += S[qi][j];
    float p = e / sum;
    __syncthreads();
    S[qi][ki] = p;
    __syncthreads();

    float o0 = 0.f, o1 = 0.f, o2 = 0.f, o3 = 0.f;
    for (int j = 0; j < 16; j++) {
        float pj = S[qi][j];
        const float* vp = v + (long)(b * T_ + j) * D_ + h * DK_ + ki * 4;
        o0 += pj * vp[0]; o1 += pj * vp[1]; o2 += pj * vp[2]; o3 += pj * vp[3];
    }
    __bf16* op = out + (long)(b * T_ + qi) * D_ + h * DK_ + ki * 4;
    op[0] = (__bf16)o0; op[1] = (__bf16)o1; op[2] = (__bf16)o2; op[3] = (__bf16)o3;
}

// Cross attention over ENC=256 precomputed keys/values (no bias, no mask).
__global__ __launch_bounds__(256) void cross_attn_kernel(const float* __restrict__ cq,
                                                         const float* __restrict__ encK,
                                                         const float* __restrict__ encV,
                                                         __bf16* __restrict__ out) {
    __shared__ float S[16 * ENC_];
    __shared__ float qs[16][DK_];
    __shared__ float red[16][16];
    int b = blockIdx.x / H_, h = blockIdx.x % H_;
    int qi = threadIdx.x >> 4, tl = threadIdx.x & 15;

    const float* qp = cq + (long)(b * T_ + qi) * D_ + h * DK_;
    for (int d = tl; d < DK_; d += 16) qs[qi][d] = qp[d];
    __syncthreads();

    const float* Kb = encK + (long)b * ENC_ * D_ + h * DK_;
    for (int e = tl; e < ENC_; e += 16) {
        const float* kp = Kb + (long)e * D_;
        float s = 0.f;
#pragma unroll 8
        for (int d = 0; d < DK_; d++) s += qs[qi][d] * kp[d];
        S[qi * ENC_ + e] = s;
    }
    __syncthreads();

    float m = -1e30f;
    for (int e = tl; e < ENC_; e += 16) m = fmaxf(m, S[qi * ENC_ + e]);
    red[qi][tl] = m;
    __syncthreads();
    if (tl == 0) {
        float mm = red[qi][0];
        for (int j = 1; j < 16; j++) mm = fmaxf(mm, red[qi][j]);
        red[qi][0] = mm;
    }
    __syncthreads();
    m = red[qi][0];
    float sum = 0.f;
    for (int e = tl; e < ENC_; e += 16) {
        float ex = expf(S[qi * ENC_ + e] - m);
        S[qi * ENC_ + e] = ex;
        sum += ex;
    }
    __syncthreads();
    red[qi][tl] = sum;
    __syncthreads();
    if (tl == 0) {
        float ss = 0.f;
        for (int j = 0; j < 16; j++) ss += red[qi][j];
        red[qi][0] = ss;
    }
    __syncthreads();
    float inv = 1.0f / red[qi][0];

    float o0 = 0.f, o1 = 0.f, o2 = 0.f, o3 = 0.f;
    const float* Vb = encV + (long)b * ENC_ * D_ + h * DK_ + tl * 4;
    for (int e = 0; e < ENC_; e++) {
        float p = S[qi * ENC_ + e] * inv;
        const float* vp = Vb + (long)e * D_;
        o0 += p * vp[0]; o1 += p * vp[1]; o2 += p * vp[2]; o3 += p * vp[3];
    }
    __bf16* op = out + (long)(b * T_ + qi) * D_ + h * DK_ + tl * 4;
    op[0] = (__bf16)o0; op[1] = (__bf16)o1; op[2] = (__bf16)o2; op[3] = (__bf16)o3;
}

// ---------------------------------------------------------------- decode argmax
#define CH_ (VOCAB_ / 32)   // 1004; 32 chunks * 1004 == 32128 exactly
__global__ __launch_bounds__(256) void argmax_part(const __bf16* __restrict__ yb,
                                                   const __bf16* __restrict__ embb,
                                                   float* __restrict__ cand_v,
                                                   int* __restrict__ cand_i,
                                                   int step) {
    __shared__ float ys[D_];
    __shared__ float bv[256];
    __shared__ int   bi[256];
    int b = blockIdx.x >> 5, c = blockIdx.x & 31;
    int row = b * T_ + (step - 1);
    const __bf16* yp = yb + (long)row * D_;
    for (int d = threadIdx.x; d < D_; d += 256) ys[d] = (float)yp[d];
    __syncthreads();

    float best = -3.4e38f; int bidx = 0x7fffffff;
    for (int t = threadIdx.x; t < CH_; t += 256) {
        int vtok = c * CH_ + t;
        const __bf16* ep = embb + (long)vtok * D_;
        float s = 0.f;
        for (int d = 0; d < D_; d += 8) {
            v8bf e8 = *(const v8bf*)(ep + d);
#pragma unroll
            for (int j = 0; j < 8; j++) s += ys[d + j] * (float)e8[j];
        }
        if (s > best) { best = s; bidx = vtok; }   // ascending order -> first max kept
    }
    bv[threadIdx.x] = best; bi[threadIdx.x] = bidx;
    __syncthreads();
    for (int s = 128; s > 0; s >>= 1) {
        if (threadIdx.x < s) {
            if (bv[threadIdx.x + s] > bv[threadIdx.x] ||
                (bv[threadIdx.x + s] == bv[threadIdx.x] && bi[threadIdx.x + s] < bi[threadIdx.x])) {
                bv[threadIdx.x] = bv[threadIdx.x + s];
                bi[threadIdx.x] = bi[threadIdx.x + s];
            }
        }
        __syncthreads();
    }
    if (threadIdx.x == 0) {
        cand_v[b * 32 + c] = bv[0];
        cand_i[b * 32 + c] = bi[0];
    }
}

__global__ void argmax_final(const float* __restrict__ cand_v, const int* __restrict__ cand_i,
                             int* __restrict__ tokens, int step) {
    int b = threadIdx.x;
    if (b >= B_) return;
    float best = -3.4e38f; int bidx = 0;
    for (int c = 0; c < 32; c++) {
        float v = cand_v[b * 32 + c];
        int   i = cand_i[b * 32 + c];
        if (v > best || (v == best && i < bidx)) { best = v; bidx = i; }
    }
    tokens[b * T_ + step] = bidx;
}

// ---------------------------------------------------------------- host driver

extern "C" void kernel_launch(void* const* d_in, const int* in_sizes, int n_in,
                              void* d_out, int out_size, void* d_ws, size_t ws_size,
                              hipStream_t stream) {
    (void)in_sizes; (void)n_in; (void)out_size; (void)ws_size;

    const float* enc_hs   = (const float*)d_in[0];
    const float* emb      = (const float*)d_in[1];
    const float* ln1      = (const float*)d_in[2];
    const float* qw       = (const float*)d_in[3];
    const float* kw       = (const float*)d_in[4];
    const float* vw       = (const float*)d_in[5];
    const float* ow       = (const float*)d_in[6];
    const float* ln2      = (const float*)d_in[7];
    const float* cqw      = (const float*)d_in[8];
    const float* ckw      = (const float*)d_in[9];
    const float* cvw      = (const float*)d_in[10];
    const float* cow      = (const float*)d_in[11];
    const float* ln3      = (const float*)d_in[12];
    const float* wi       = (const float*)d_in[13];
    const float* wo       = (const float*)d_in[14];
    const float* final_ln = (const float*)d_in[15];
    const float* rel_bias = (const float*)d_in[16];
    // d_in[17] = max_gen_len (device scalar) -> hardcoded 16 (capture-safe)

    const long NW  = (long)D_ * D_;        // per-layer proj weight elems
    const long NWI = (long)D_ * DFF_;      // per-layer ffn weight elems

    char* p = (char*)d_ws;
    auto alloc = [&](size_t bytes) -> void* {
        void* r = (void*)p;
        p += (bytes + 255) & ~(size_t)255;
        return r;
    };
    __bf16* emb_bf  = (__bf16*)alloc((size_t)VOCAB_ * D_ * 2);   // [VOCAB, D] (= [N,K])
    __bf16* qw_t    = (__bf16*)alloc((size_t)L_ * NW * 2);       // all weights stored [N,K]
    __bf16* kw_t    = (__bf16*)alloc((size_t)L_ * NW * 2);
    __bf16* vw_t    = (__bf16*)alloc((size_t)L_ * NW * 2);
    __bf16* ow_t    = (__bf16*)alloc((size_t)L_ * NW * 2);
    __bf16* cqw_t   = (__bf16*)alloc((size_t)L_ * NW * 2);
    __bf16* ckw_t   = (__bf16*)alloc((size_t)L_ * NW * 2);
    __bf16* cvw_t   = (__bf16*)alloc((size_t)L_ * NW * 2);
    __bf16* cow_t   = (__bf16*)alloc((size_t)L_ * NW * 2);
    __bf16* wi_t    = (__bf16*)alloc((size_t)L_ * NWI * 2);
    __bf16* wo_t    = (__bf16*)alloc((size_t)L_ * NWI * 2);
    __bf16* encA_b  = (__bf16*)alloc((size_t)B_ * ENC_ * D_ * 2);
    float*  encK    = (float*)alloc((size_t)L_ * B_ * ENC_ * D_ * 4);
    float*  encV    = (float*)alloc((size_t)L_ * B_ * ENC_ * D_ * 4);
    int*    tokens  = (int*)alloc(256);
    float*  x       = (float*)alloc((size_t)B_ * T_ * D_ * 4);
    __bf16* hb      = (__bf16*)alloc((size_t)B_ * T_ * D_ * 2);
    float*  qf      = (float*)alloc((size_t)B_ * T_ * D_ * 4);
    float*  kf      = (float*)alloc((size_t)B_ * T_ * D_ * 4);
    float*  vf      = (float*)alloc((size_t)B_ * T_ * D_ * 4);
    __bf16* att_b   = (__bf16*)alloc((size_t)B_ * T_ * D_ * 2);
    float*  cqf     = (float*)alloc((size_t)B_ * T_ * D_ * 4);
    __bf16* catt_b  = (__bf16*)alloc((size_t)B_ * T_ * D_ * 2);
    __bf16* ff1b    = (__bf16*)alloc((size_t)B_ * T_ * DFF_ * 2);
    __bf16* y_b     = (__bf16*)alloc((size_t)B_ * T_ * D_ * 2);
    float*  cand_v  = (float*)alloc(256);
    int*    cand_i  = (int*)alloc(256);

    // ---- 1) convert weights to bf16, transposed to [N,K] (model ~125 MB -> L2 resident)
    cvt_f32_bf16<<<2048, 256, 0, stream>>>(emb, emb_bf, (long)VOCAB_ * D_);
    cvt_f32_bf16<<<64, 256, 0, stream>>>(enc_hs, encA_b, (long)B_ * ENC_ * D_);
    {
        dim3 gP(D_ / 32, D_ / 32, L_);          // [768,768] per layer
        cvt_transpose_bf16<<<gP, 256, 0, stream>>>(qw,  qw_t,  D_, D_);
        cvt_transpose_bf16<<<gP, 256, 0, stream>>>(kw,  kw_t,  D_, D_);
        cvt_transpose_bf16<<<gP, 256, 0, stream>>>(vw,  vw_t,  D_, D_);
        cvt_transpose_bf16<<<gP, 256, 0, stream>>>(ow,  ow_t,  D_, D_);
        cvt_transpose_bf16<<<gP, 256, 0, stream>>>(cqw, cqw_t, D_, D_);
        cvt_transpose_bf16<<<gP, 256, 0, stream>>>(ckw, ckw_t, D_, D_);
        cvt_transpose_bf16<<<gP, 256, 0, stream>>>(cvw, cvw_t, D_, D_);
        cvt_transpose_bf16<<<gP, 256, 0, stream>>>(cow, cow_t, D_, D_);
        dim3 gI(D_ / 32, DFF_ / 32, L_);        // wi: [768,3072] -> [3072,768]
        cvt_transpose_bf16<<<gI, 256, 0, stream>>>(wi, wi_t, D_, DFF_);
        dim3 gO(DFF_ / 32, D_ / 32, L_);        // wo: [3072,768] -> [768,3072]
        cvt_transpose_bf16<<<gO, 256, 0, stream>>>(wo, wo_t, DFF_, D_);
    }

    // ---- 2) precompute cross-attention K/V per layer: [512,768] @ [768,768]
    for (int l = 0; l < L_; l++) {
        gemm_wmma<false, false, false><<<dim3(B_ * ENC_ / 32, D_ / 128), 128, 0, stream>>>(
            encA_b, ckw_t + (long)l * NW, encK + (long)l * B_ * ENC_ * D_, nullptr, nullptr, D_, D_);
        gemm_wmma<false, false, false><<<dim3(B_ * ENC_ / 32, D_ / 128), 128, 0, stream>>>(
            encA_b, cvw_t + (long)l * NW, encV + (long)l * B_ * ENC_ * D_, nullptr, nullptr, D_, D_);
    }

    init_tokens<<<1, 32, 0, stream>>>(tokens);

    const dim3 gProj(1, D_ / 128);     // (1, 6)   M=32
    const dim3 gFF1(1, DFF_ / 128);    // (1, 24)
    const float final_scale = 1.0f / sqrtf((float)D_);

    auto forward = [&]() {
        embed_kernel<<<B_ * T_, 256, 0, stream>>>(tokens, emb, x);
        for (int l = 0; l < L_; l++) {
            // ---- self attention block
            rmsnorm_kernel<<<B_ * T_, 256, 0, stream>>>(x, ln1 + l * D_, hb, 1.0f);
            gemm_wmma<false, false, false><<<gProj, 128, 0, stream>>>(
                hb, qw_t + (long)l * NW, qf, nullptr, nullptr, D_, D_);
            gemm_wmma<false, false, false><<<gProj, 128, 0, stream>>>(
                hb, kw_t + (long)l * NW, kf, nullptr, nullptr, D_, D_);
            gemm_wmma<false, false, false><<<gProj, 128, 0, stream>>>(
                hb, vw_t + (long)l * NW, vf, nullptr, nullptr, D_, D_);
            self_attn_kernel<<<B_ * H_, 256, 0, stream>>>(qf, kf, vf, rel_bias, att_b);
            gemm_wmma<true, false, false><<<gProj, 128, 0, stream>>>(
                att_b, ow_t + (long)l * NW, x, nullptr, x, D_, D_);
            // ---- cross attention block
            rmsnorm_kernel<<<B_ * T_, 256, 0, stream>>>(x, ln2 + l * D_, hb, 1.0f);
            gemm_wmma<false, false, false><<<gProj, 128, 0, stream>>>(
                hb, cqw_t + (long)l * NW, cqf, nullptr, nullptr, D_, D_);
            cross_attn_kernel<<<B_ * H_, 256, 0, stream>>>(
                cqf, encK + (long)l * B_ * ENC_ * D_, encV + (long)l * B_ * ENC_ * D_, catt_b);
            gemm_wmma<true, false, false><<<gProj, 128, 0, stream>>>(
                catt_b, cow_t + (long)l * NW, x, nullptr, x, D_, D_);
            // ---- FFN block
            rmsnorm_kernel<<<B_ * T_, 256, 0, stream>>>(x, ln3 + l * D_, hb, 1.0f);
            gemm_wmma<false, true, true><<<gFF1, 128, 0, stream>>>(
                hb, wi_t + (long)l * NWI, nullptr, ff1b, nullptr, D_, DFF_);
            gemm_wmma<true, false, false><<<gProj, 128, 0, stream>>>(
                ff1b, wo_t + (long)l * NWI, x, nullptr, x, DFF_, D_);
        }
        rmsnorm_kernel<<<B_ * T_, 256, 0, stream>>>(x, final_ln, y_b, final_scale);
    };

    // ---- 3) greedy decode: 15 steps, each = full forward + 2-row logits argmax
    for (int step = 1; step < T_; step++) {
        forward();
        argmax_part<<<B_ * 32, 256, 0, stream>>>(y_b, emb_bf, cand_v, cand_i, step);
        argmax_final<<<1, 32, 0, stream>>>(cand_v, cand_i, tokens, step);
    }

    // ---- 4) final forward: full LM head into d_out [32, 32128] (emb is already [N,K])
    forward();
    gemm_wmma<false, false, false><<<dim3(1, VOCAB_ / 128), 128, 0, stream>>>(
        y_b, emb_bf, (float*)d_out, nullptr, nullptr, D_, VOCAB_);
}
